// SpatioTemporalAGCLSTM_67860483277064
// MI455X (gfx1250) — matrix operand order repaired and checked by reading, not compiled, not run on hardware
//
#include <hip/hip_runtime.h>
#include <hip/hip_bf16.h>
#include <math.h>

// ---------------- problem constants (match reference) ----------------
#define BB   8
#define TT   24
#define NN   10000
#define FF   16
#define HH   64
#define EE   160000
#define OWW  12
#define E2   (EE + NN)          // edges + self loops
#define TBH  (TT * BB * HH)     // 12288

typedef __attribute__((ext_vector_type(16))) _Float16 v16h;
typedef __attribute__((ext_vector_type(8)))  float    v8f;

// ---------------- workspace layout (floats) ----------------
// deg/dinv : NN
// norm     : E2
// xw       : BB*NN*HH   (per current timestep)
// agg      : BB*NN*HH
// rnn_in   : TT*BB*HH
#define WS_DEG    0
#define WS_NORM   (WS_DEG + NN)
#define WS_XW     (WS_NORM + E2)
#define WS_AGG    (WS_XW + BB * NN * HH)
#define WS_RNNIN  (WS_AGG + BB * NN * HH)

// ============================================================
// 0) zero deg + rnn_in (accumulators we atomically add into)
// ============================================================
__global__ void k_init(float* __restrict__ ws) {
    int i = blockIdx.x * blockDim.x + threadIdx.x;
    if (i < NN)  ws[WS_DEG + i] = 0.0f;
    if (i < TBH) ws[WS_RNNIN + i] = 0.0f;
}

// ============================================================
// 1) degree scatter:  deg[dst] += ew  (self-loop "+1" folded later)
// ============================================================
__global__ void k_deg(const int* __restrict__ ei, const float* __restrict__ ew,
                      float* __restrict__ ws) {
    int e = blockIdx.x * blockDim.x + threadIdx.x;
    if (e < EE) {
        int dst = ei[EE + e];
        atomicAdd(&ws[WS_DEG + dst], ew[e]);
    }
}

// ============================================================
// 2) dinv = rsqrt(deg + 1)  (in place; deg+1 > 0 always)
// ============================================================
__global__ void k_dinv(float* __restrict__ ws) {
    int i = blockIdx.x * blockDim.x + threadIdx.x;
    if (i < NN) ws[WS_DEG + i] = rsqrtf(ws[WS_DEG + i] + 1.0f);
}

// ============================================================
// 3) norm[e] = dinv[src]*ew*dinv[dst]; self-loops: dinv[i]^2
// ============================================================
__global__ void k_norm(const int* __restrict__ ei, const float* __restrict__ ew,
                       float* __restrict__ ws) {
    int e = blockIdx.x * blockDim.x + threadIdx.x;
    if (e >= E2) return;
    const float* dinv = ws + WS_DEG;
    float nv;
    if (e < EE) {
        int src = ei[e], dst = ei[EE + e];
        nv = dinv[src] * ew[e] * dinv[dst];
    } else {
        int i = e - EE;
        nv = dinv[i] * dinv[i];
    }
    ws[WS_NORM + e] = nv;
}

// ============================================================
// 4) zero agg (float4 stores)
// ============================================================
__global__ void k_zero_agg(float* __restrict__ ws) {
    int i = blockIdx.x * blockDim.x + threadIdx.x;  // counts float4s
    float4* p = (float4*)(ws + WS_AGG);
    if (i < (BB * NN * HH) / 4) p[i] = make_float4(0.f, 0.f, 0.f, 0.f);
}

// ============================================================
// 5) WMMA GEMM:  xw[b,n,:] = x[b,t,n,:] @ W_gcn    (f16 in, f32 acc)
//    one wave (32 lanes) per 16-node tile of one batch; 4 H-tiles,
//    A-fragment reused across the 4 v_wmma_f32_16x16x32_f16 ops.
//    K = F = 16 padded to 32 with zeros.
// ============================================================
__global__ __launch_bounds__(32)
void k_gcn_gemm(const float* __restrict__ x, const float* __restrict__ Wg,
                float* __restrict__ ws, int t) {
    const int lane  = threadIdx.x;        // 0..31
    const int m     = lane & 15;          // row within tile / N column for B,C
    const int grp   = lane >> 4;          // half-wave group
    const int ntile = blockIdx.x;         // 0..624  (625*16 == 10000 exactly)
    const int b     = blockIdx.y;
    const int node  = ntile * 16 + m;

    // ---- A fragment: 16x32 f16 (K 16..31 zero-padded) ----
    // lanes 0-15: halves 0..7 = K 0..7 ; lanes 16-31: halves 0..7 = K 8..15
    const float* xp = x + (((size_t)b * TT + t) * NN + node) * FF;
    const float4* xq = (const float4*)xp;
    float4 v0 = xq[grp * 2 + 0];
    float4 v1 = xq[grp * 2 + 1];
    v16h a;
    a[0] = (_Float16)v0.x; a[1] = (_Float16)v0.y;
    a[2] = (_Float16)v0.z; a[3] = (_Float16)v0.w;
    a[4] = (_Float16)v1.x; a[5] = (_Float16)v1.y;
    a[6] = (_Float16)v1.z; a[7] = (_Float16)v1.w;
#pragma unroll
    for (int i = 8; i < 16; ++i) a[i] = (_Float16)0.0f;

    float* xw = ws + WS_XW;

#pragma unroll
    for (int ht = 0; ht < 4; ++ht) {
        const int col = ht * 16 + m;      // output column in H
        // ---- B fragment: 32x16 f16. lanes 0-15 halves 0..7 = K 0..7, lanes 16-31
        //      halves 0..7 = K 8..15; halves 8..15 = K 16..31 (zero padding). ----
        v16h bm;
#pragma unroll
        for (int k = 0; k < 8; ++k) bm[k] = (_Float16)Wg[(grp * 8 + k) * HH + col];
#pragma unroll
        for (int k = 8; k < 16; ++k) bm[k] = (_Float16)0.0f;
        v8f c = {};
        c = __builtin_amdgcn_wmma_f32_16x16x32_f16(
                /*neg_a=*/false, a, /*neg_b=*/false, bm,
                /*c_mod=*/(short)0, c, /*reuse_a=*/false, /*reuse_b=*/false);
        // ---- store C: VGPR r -> M = r (lanes 0-15) / r+8 (lanes 16-31) ----
#pragma unroll
        for (int r = 0; r < 8; ++r) {
            int mm = r + grp * 8;
            xw[((size_t)b * NN + ntile * 16 + mm) * HH + col] = c[r];
        }
    }
}

// ============================================================
// 6) edge scatter: agg[b,dst,h] += norm[e] * xw[b,src,h]
//    thread = (b from grid.y) x (edge, h-quad); float4 load, 4 atomics
// ============================================================
__global__ void k_scatter(const int* __restrict__ ei, float* __restrict__ ws) {
    int idx = blockIdx.x * blockDim.x + threadIdx.x;   // e*16 + hq
    if (idx >= E2 * 16) return;
    int e  = idx >> 4;
    int h  = (idx & 15) * 4;
    int b  = blockIdx.y;

    int src, dst;
    if (e < EE) { src = ei[e]; dst = ei[EE + e]; }
    else        { src = dst = e - EE; }

    float nv = ws[WS_NORM + e];
    const float4 xv = *(const float4*)(ws + WS_XW + ((size_t)b * NN + src) * HH + h);
    float* ag = ws + WS_AGG + ((size_t)b * NN + dst) * HH + h;
    atomicAdd(ag + 0, nv * xv.x);
    atomicAdd(ag + 1, nv * xv.y);
    atomicAdd(ag + 2, nv * xv.z);
    atomicAdd(ag + 3, nv * xv.w);
}

// ============================================================
// 7) node mean of relu(agg + b_gcn) -> rnn_in[t,b,:]
//    grid (chunks=40, B), block 64 (one thread per h), 250 nodes/chunk
// ============================================================
__global__ __launch_bounds__(64)
void k_reduce(const float* __restrict__ bg, float* __restrict__ ws, int t) {
    const int h  = threadIdx.x;
    const int b  = blockIdx.y;
    const int n0 = blockIdx.x * 250;
    const float bias = bg[h];
    const float* ag = ws + WS_AGG + (size_t)b * NN * HH;
    float acc = 0.0f;
    for (int n = n0; n < n0 + 250; ++n) {
        float v = ag[(size_t)n * HH + h] + bias;
        acc += fmaxf(v, 0.0f);
    }
    atomicAdd(&ws[WS_RNNIN + ((size_t)t * BB + b) * HH + h], acc * (1.0f / NN));
}

// ============================================================
// 8) GRU (T sequential steps) + attention + softmax + FC
//    one workgroup of 512 threads (= B*H), everything in LDS.
// ============================================================
__device__ __forceinline__ float sigmoidf_(float x) { return 1.0f / (1.0f + expf(-x)); }

__global__ __launch_bounds__(512)
void k_gru_attn(const float* __restrict__ ws_ro,
                const float* __restrict__ W_ih, const float* __restrict__ W_hh,
                const float* __restrict__ b_ih, const float* __restrict__ b_hh,
                const float* __restrict__ W_attn, const float* __restrict__ b_attn,
                const float* __restrict__ v_w,
                const float* __restrict__ W_fc, const float* __restrict__ b_fc,
                float* __restrict__ out) {
    __shared__ float rnn_sh[TT][BB][HH];   // 48 KB
    __shared__ float hbuf[BB][HH];
    __shared__ float xsh[BB][HH];
    __shared__ float ctx[BB][HH];
    __shared__ float sc_sh[BB][TT];
    __shared__ float aw_sh[BB][TT];

    const int tid = threadIdx.x;
    const int b   = tid >> 6;
    const int h   = tid & 63;
    const float* rnn_in = ws_ro + WS_RNNIN;

    hbuf[b][h] = 0.0f;
    __syncthreads();

    // ---------------- GRU over T (torch gate order r|z|n) ----------------
    for (int t = 0; t < TT; ++t) {
        xsh[b][h] = rnn_in[((size_t)t * BB + b) * HH + h];
        __syncthreads();
        float gr = b_ih[h], gz = b_ih[HH + h], gn = b_ih[2 * HH + h];
        float hr = b_hh[h], hz = b_hh[HH + h], hn = b_hh[2 * HH + h];
        for (int k = 0; k < HH; ++k) {
            const float xk = xsh[b][k];
            const float hk = hbuf[b][k];
            const float* wi = W_ih + k * 3 * HH;
            const float* wh = W_hh + k * 3 * HH;
            gr += xk * wi[h];            gz += xk * wi[HH + h];
            gn += xk * wi[2 * HH + h];
            hr += hk * wh[h];            hz += hk * wh[HH + h];
            hn += hk * wh[2 * HH + h];
        }
        const float r = sigmoidf_(gr + hr);
        const float z = sigmoidf_(gz + hz);
        const float n = tanhf(gn + r * hn);
        const float hold = hbuf[b][h];
        const float hnew = (1.0f - z) * n + z * hold;
        __syncthreads();
        hbuf[b][h]      = hnew;
        rnn_sh[t][b][h] = hnew;
        __syncthreads();
    }

    // ---------------- attention scores ----------------
    if (tid < BB * TT) {
        const int bb = tid / TT, t = tid % TT;
        float s = 0.0f;
        for (int j = 0; j < HH; ++j) {
            float e = b_attn[j];
            for (int k = 0; k < HH; ++k) {
                e += hbuf[bb][k]      * W_attn[k * HH + j];
                e += rnn_sh[t][bb][k] * W_attn[(HH + k) * HH + j];
            }
            s += tanhf(e) * v_w[j];
        }
        sc_sh[bb][t] = s;
    }
    __syncthreads();

    // ---------------- softmax over T (one thread per batch) ----------------
    if (tid < BB) {
        float mx = -1e30f;
        for (int t = 0; t < TT; ++t) mx = fmaxf(mx, sc_sh[tid][t]);
        float sum = 0.0f;
        for (int t = 0; t < TT; ++t) { float e = expf(sc_sh[tid][t] - mx); aw_sh[tid][t] = e; sum += e; }
        const float inv = 1.0f / sum;
        for (int t = 0; t < TT; ++t) {
            const float w = aw_sh[tid][t] * inv;
            aw_sh[tid][t] = w;
            out[BB * OWW + tid * TT + t] = w;   // attn_w output (second tuple elem)
        }
    }
    __syncthreads();

    // ---------------- context = attn_w @ rnn_outputs ----------------
    float c = 0.0f;
    for (int t = 0; t < TT; ++t) c += aw_sh[b][t] * rnn_sh[t][b][h];
    ctx[b][h] = c;
    __syncthreads();

    // ---------------- fc: [last_hidden | context] @ W_fc + b_fc ----------------
    if (tid < BB * OWW) {
        const int bb = tid / OWW, ow = tid % OWW;
        float val = b_fc[ow];
        for (int k = 0; k < HH; ++k) {
            val += rnn_sh[TT - 1][bb][k] * W_fc[k * OWW + ow];
            val += ctx[bb][k]            * W_fc[(HH + k) * OWW + ow];
        }
        out[bb * OWW + ow] = val;
    }
}

// ============================================================
// host-side launcher
// ============================================================
extern "C" void kernel_launch(void* const* d_in, const int* in_sizes, int n_in,
                              void* d_out, int out_size, void* d_ws, size_t ws_size,
                              hipStream_t stream) {
    const float* x      = (const float*)d_in[0];
    const int*   ei     = (const int*)  d_in[1];
    const float* ew     = (const float*)d_in[2];
    const float* W_gcn  = (const float*)d_in[3];
    const float* b_gcn  = (const float*)d_in[4];
    const float* W_ih   = (const float*)d_in[5];
    const float* W_hh   = (const float*)d_in[6];
    const float* b_ih   = (const float*)d_in[7];
    const float* b_hh   = (const float*)d_in[8];
    const float* W_attn = (const float*)d_in[9];
    const float* b_attn = (const float*)d_in[10];
    const float* v_w    = (const float*)d_in[11];
    const float* W_fc   = (const float*)d_in[12];
    const float* b_fc   = (const float*)d_in[13];
    float* out = (float*)d_out;
    float* ws  = (float*)d_ws;

    // ---- graph normalization (once) ----
    k_init<<<(TBH + 255) / 256, 256, 0, stream>>>(ws);
    k_deg <<<(EE + 255) / 256, 256, 0, stream>>>(ei, ew, ws);
    k_dinv<<<(NN + 255) / 256, 256, 0, stream>>>(ws);
    k_norm<<<(E2 + 255) / 256, 256, 0, stream>>>(ei, ew, ws);

    // ---- per-timestep GCN: WMMA gemm -> edge scatter -> node mean ----
    const dim3 gemm_grid(NN / 16, BB);          // 625 x 8 waves
    const dim3 scat_grid((E2 * 16 + 255) / 256, BB);
    const dim3 red_grid(NN / 250, BB);          // 40 x 8
    for (int t = 0; t < TT; ++t) {
        k_zero_agg<<<(BB * NN * HH / 4 + 255) / 256, 256, 0, stream>>>(ws);
        k_gcn_gemm<<<gemm_grid, 32, 0, stream>>>(x, W_gcn, ws, t);
        k_scatter <<<scat_grid, 256, 0, stream>>>(ei, ws);
        k_reduce  <<<red_grid, 64, 0, stream>>>(b_gcn, ws, t);
    }

    // ---- sequential GRU + attention + FC in one WGP-resident workgroup ----
    k_gru_attn<<<1, 512, 0, stream>>>(ws, W_ih, W_hh, b_ih, b_hh,
                                      W_attn, b_attn, v_w, W_fc, b_fc, out);
}
